// AttentionBlock_28467043238644
// MI455X (gfx1250) — matrix operand order repaired and checked by reading, not compiled, not run on hardware
//
#include <hip/hip_runtime.h>
#include <hip/hip_bf16.h>
#include <math.h>

// ---------------------------------------------------------------------------
// Types for CDNA5 WMMA (gfx1250, wave32)
// ---------------------------------------------------------------------------
typedef __attribute__((ext_vector_type(16))) __bf16          v16bf;
typedef __attribute__((ext_vector_type(8)))  float           v8f;
typedef __attribute__((ext_vector_type(8)))  unsigned short  v8u16;

union Frag {            // 32 bytes = 8 VGPRs = one 16x32 bf16 A or 32x16 bf16 B fragment
    v16bf  bf;
    v8u16  h[2];
};

static __device__ __forceinline__ unsigned short f32_to_bf16(float f) {
    unsigned int u = __float_as_uint(f);
    unsigned int r = u + 0x7FFFu + ((u >> 16) & 1u);   // round-to-nearest-even
    return (unsigned short)(r >> 16);
}
static __device__ __forceinline__ float eluf(float v) {
    return v > 0.0f ? v : (expf(v) - 1.0f);
}
static __device__ __forceinline__ float sigmf(float v) {
    return 1.0f / (1.0f + expf(-v));
}

// Problem constants
#define NB   32
#define HW   1024
#define PTOT 32768            // NB * HW, global column count
#define BC   4                // attention batch chunk

// ---------------------------------------------------------------------------
// Generic WMMA GEMM:  Out[O][P] = Wb[Opad][Cpad] (bf16) x Act (bf16, B-layout)
//   Act B-layout: elem (c,p) at ((c>>5)*P + p)*32 + (c&31)
//   mode 0: f32 row-major out [r*P + col] (+bias, optional accumulate)
//   mode 1: bf16 B-layout out
//   mode 2: bf16 row-major out
// block (32,8): 8 waves, each wave = 16(M) x 64(N) tile, K step 32.
// All 8 waves share the same 64-column B panel -> stage it in LDS once per
// K-step (double buffered); WMMAs read from ds_load_b128 while the next
// panel's global_load_b128 is in flight.
// ---------------------------------------------------------------------------
__global__ __launch_bounds__(256)
void wmma_gemm(const unsigned short* __restrict__ Wb,
               const unsigned short* __restrict__ Act,
               const float* __restrict__ bias,
               float* __restrict__ OutF,
               unsigned short* __restrict__ OutB,
               int Oreal, int Opad, int Cpad, int P, int mode, int accum) {
    __shared__ __align__(16) unsigned short sB[2][2048];   // 2 x 32x64 bf16 panels

    const int lane = threadIdx.x;
    const int tid  = threadIdx.y * 32 + lane;
    const int mt   = blockIdx.y * blockDim.y + threadIdx.y;
    const bool active = (mt * 16 < Opad);
    const int m0   = active ? mt * 16 : 0;                 // clamp: keep waves in barriers
    const int p0   = blockIdx.x * 64;
    const int half = lane >> 4;          // 0: lanes 0-15, 1: lanes 16-31
    const int l16  = lane & 15;

    const unsigned short* wrow = Wb + (size_t)(m0 + l16) * Cpad;
    // stage first B panel: 4KB panel, 256 threads x 16B
    {
        v8u16 s0 = *(const v8u16*)(Act + (size_t)p0 * 32 + tid * 8);
        *(v8u16*)(&sB[0][tid * 8]) = s0;
    }
    Frag a;
    a.h[0] = *(const v8u16*)(wrow + half * 8);
    a.h[1] = *(const v8u16*)(wrow + half * 8 + 16);
    __syncthreads();

    v8f acc0 = {}, acc1 = {}, acc2 = {}, acc3 = {};

    for (int k = 0; k < Cpad; k += 32) {
        const int buf = (k >> 5) & 1;
        const bool hasNext = (k + 32 < Cpad);

        v8u16 nxtPanel;
        Frag  an;
        if (hasNext) {
            nxtPanel = *(const v8u16*)(Act + ((size_t)((k + 32) >> 5) * P + p0) * 32 + tid * 8);
            const unsigned short* apn = wrow + (k + 32) + half * 8;
            an.h[0] = *(const v8u16*)(apn);
            an.h[1] = *(const v8u16*)(apn + 16);
            if (k + 64 < Cpad)
                __builtin_prefetch(Act + ((size_t)((k + 64) >> 5) * P + p0) * 32 + tid * 8, 0, 1);
        }

        const unsigned short* sb = &sB[buf][(size_t)l16 * 32 + half * 8];
        Frag b0, b1, b2, b3;
        b0.h[0] = *(const v8u16*)(sb);
        b0.h[1] = *(const v8u16*)(sb + 16);
        b1.h[0] = *(const v8u16*)(sb + 16 * 32);
        b1.h[1] = *(const v8u16*)(sb + 16 * 32 + 16);
        b2.h[0] = *(const v8u16*)(sb + 32 * 32);
        b2.h[1] = *(const v8u16*)(sb + 32 * 32 + 16);
        b3.h[0] = *(const v8u16*)(sb + 48 * 32);
        b3.h[1] = *(const v8u16*)(sb + 48 * 32 + 16);

        acc0 = __builtin_amdgcn_wmma_f32_16x16x32_bf16(false, a.bf, false, b0.bf, (short)0, acc0, false, false);
        acc1 = __builtin_amdgcn_wmma_f32_16x16x32_bf16(false, a.bf, false, b1.bf, (short)0, acc1, false, false);
        acc2 = __builtin_amdgcn_wmma_f32_16x16x32_bf16(false, a.bf, false, b2.bf, (short)0, acc2, false, false);
        acc3 = __builtin_amdgcn_wmma_f32_16x16x32_bf16(false, a.bf, false, b3.bf, (short)0, acc3, false, false);

        if (hasNext) {
            __syncthreads();                     // everyone done reading sB[buf^1]? (prev round)
            *(v8u16*)(&sB[buf ^ 1][tid * 8]) = nxtPanel;
            a = an;
            __syncthreads();                     // next panel visible
        }
    }

    if (!active) return;

    v8f accs[4] = {acc0, acc1, acc2, acc3};
#pragma unroll
    for (int nt = 0; nt < 4; ++nt) {
        const int col = p0 + nt * 16 + l16;
#pragma unroll
        for (int v = 0; v < 8; ++v) {
            const int r = m0 + v + half * 8;
            if (r < Oreal) {
                float val = accs[nt][v] + (bias ? bias[r] : 0.0f);
                if (mode == 0) {
                    size_t o = (size_t)r * P + col;
                    OutF[o] = accum ? (OutF[o] + val) : val;
                } else if (mode == 1) {
                    OutB[((size_t)(r >> 5) * P + col) * 32 + (r & 31)] = f32_to_bf16(val);
                } else {
                    OutB[(size_t)r * P + col] = f32_to_bf16(val);
                }
            }
        }
    }
}

// ---------------------------------------------------------------------------
// QK^T: S[q,kk] = sum_ch Q[ch,q]*K[ch,kk]. Q,K stored B-layout [32][PTOT]
// (ch rows 16..31 zero). One WMMA per 16x16 tile (K=32). Per chunk of BC batches.
// block (32,8): 8 waves x 16 q-rows; grid (16 kk-groups, 8 q-groups, BC)
// ---------------------------------------------------------------------------
__global__ __launch_bounds__(256)
void qk_gemm(const unsigned short* __restrict__ Qb,
             const unsigned short* __restrict__ Kb,
             float* __restrict__ S, int batch0) {
    const int lane = threadIdx.x;
    const int half = lane >> 4;
    const int l16  = lane & 15;
    const int bz   = blockIdx.z;
    const int bg   = batch0 + bz;
    const int q0   = blockIdx.y * 128 + threadIdx.y * 16;
    const int k0   = blockIdx.x * 64;

    Frag a;
    const unsigned short* qp = Qb + (size_t)(bg * HW + q0 + l16) * 32 + half * 8;
    a.h[0] = *(const v8u16*)(qp);
    a.h[1] = *(const v8u16*)(qp + 16);

    Frag b[4];
#pragma unroll
    for (int nt = 0; nt < 4; ++nt) {
        const unsigned short* kp = Kb + (size_t)(bg * HW + k0 + nt * 16 + l16) * 32 + half * 8;
        b[nt].h[0] = *(const v8u16*)(kp);
        b[nt].h[1] = *(const v8u16*)(kp + 16);
    }
    v8f acc[4] = {};
#pragma unroll
    for (int nt = 0; nt < 4; ++nt)
        acc[nt] = __builtin_amdgcn_wmma_f32_16x16x32_bf16(false, a.bf, false, b[nt].bf, (short)0, acc[nt], false, false);

#pragma unroll
    for (int nt = 0; nt < 4; ++nt)
#pragma unroll
        for (int v = 0; v < 8; ++v) {
            const int qr = q0 + v + half * 8;
            S[((size_t)(bz * HW + qr)) * HW + k0 + nt * 16 + l16] = acc[nt][v];
        }
}

// ---------------------------------------------------------------------------
// Faithful causal softmax: mask strictly-lower (col<row), softmax over full
// row including the injected zeros, re-mask, renormalize by masked sum + eps.
// Emits bf16 weights directly in AV B-layout: wb[((j>>5)*HW + q)*32 + (j&31)].
// One wave per row. block (32,8); grid (128, BC)
// ---------------------------------------------------------------------------
__global__ __launch_bounds__(256)
void softmax_rows(const float* __restrict__ S, unsigned short* __restrict__ Wb) {
    const int lane = threadIdx.x;
    const int q    = blockIdx.x * 8 + threadIdx.y;
    const int bz   = blockIdx.y;
    const float* srow = S + ((size_t)(bz * HW + q)) * HW;

    float mx = 0.0f;                                  // masked entries are literal 0
    for (int it = 0; it < 32; ++it) {
        int c = it * 32 + lane;
        float v = (c < q) ? srow[c] : 0.0f;
        mx = fmaxf(mx, v);
    }
#pragma unroll
    for (int off = 16; off; off >>= 1) mx = fmaxf(mx, __shfl_xor(mx, off, 32));

    float A = 0.0f, D = 0.0f;                         // A: masked sum, D: full-row sum
    for (int it = 0; it < 32; ++it) {
        int c = it * 32 + lane;
        float v = (c < q) ? srow[c] : 0.0f;
        float e = __expf(v - mx);
        D += e;
        if (c < q) A += e;
    }
#pragma unroll
    for (int off = 16; off; off >>= 1) { A += __shfl_xor(A, off, 32); D += __shfl_xor(D, off, 32); }

    const float inv = 1.0f / (A + 1e-7f * D);         // == softmax / (masked_sum + eps)
    unsigned short* wrow = Wb + (size_t)bz * HW * HW;
    for (int it = 0; it < 32; ++it) {
        int c = it * 32 + lane;
        float w = (c < q) ? __expf(srow[c] - mx) * inv : 0.0f;
        wrow[((size_t)it * HW + q) * 32 + lane] = f32_to_bf16(w);
    }
}

// ---------------------------------------------------------------------------
// AV: att_v[c,q] = sum_j V[c,j]*W[q,j].  V bf16 row-major [80][PTOT] (A-op),
// W bf16 B-layout per batch (B-op). One wave per 16x64 tile, K loop over 1024,
// register double-buffered so next-step loads overlap current WMMAs.
// grid (16 q-groups, 5 c-tiles, BC); block 32
// ---------------------------------------------------------------------------
__global__ __launch_bounds__(32)
void av_gemm(const unsigned short* __restrict__ Vrow,
             const unsigned short* __restrict__ Wb,
             float* __restrict__ AttV, int batch0) {
    const int lane = threadIdx.x;
    const int half = lane >> 4;
    const int l16  = lane & 15;
    const int bz   = blockIdx.z;
    const int bg   = batch0 + bz;
    const int c0   = blockIdx.y * 16;
    const int q0   = blockIdx.x * 64;

    const unsigned short* vp0 = Vrow + (size_t)(c0 + l16) * PTOT + (size_t)bg * HW;
    const unsigned short* wb0 = Wb + (size_t)bz * HW * HW;

    Frag a, b[4];
    a.h[0] = *(const v8u16*)(vp0 + half * 8);
    a.h[1] = *(const v8u16*)(vp0 + half * 8 + 16);
    {
        const size_t bbase = ((size_t)q0 + l16) * 32 + half * 8;
#pragma unroll
        for (int nt = 0; nt < 4; ++nt) {
            b[nt].h[0] = *(const v8u16*)(wb0 + bbase + nt * 16 * 32);
            b[nt].h[1] = *(const v8u16*)(wb0 + bbase + nt * 16 * 32 + 16);
        }
    }

    v8f acc[4] = {};
    for (int k = 0; k < HW; k += 32) {
        Frag an, bn[4];
        const bool hasNext = (k + 32 < HW);
        if (hasNext) {
            an.h[0] = *(const v8u16*)(vp0 + (k + 32) + half * 8);
            an.h[1] = *(const v8u16*)(vp0 + (k + 32) + half * 8 + 16);
            const size_t bbase = ((size_t)((k + 32) >> 5) * HW + q0 + l16) * 32 + half * 8;
#pragma unroll
            for (int nt = 0; nt < 4; ++nt) {
                bn[nt].h[0] = *(const v8u16*)(wb0 + bbase + nt * 16 * 32);
                bn[nt].h[1] = *(const v8u16*)(wb0 + bbase + nt * 16 * 32 + 16);
            }
        }
#pragma unroll
        for (int nt = 0; nt < 4; ++nt)
            acc[nt] = __builtin_amdgcn_wmma_f32_16x16x32_bf16(false, a.bf, false, b[nt].bf, (short)0, acc[nt], false, false);
        if (hasNext) {
            a = an;
#pragma unroll
            for (int nt = 0; nt < 4; ++nt) b[nt] = bn[nt];
        }
    }
#pragma unroll
    for (int nt = 0; nt < 4; ++nt)
#pragma unroll
        for (int v = 0; v < 8; ++v)
            AttV[(size_t)(c0 + v + half * 8) * PTOT + (size_t)bg * HW + q0 + nt * 16 + l16] = acc[nt][v];
}

// ---------------------------------------------------------------------------
// Element-wise staging kernels
// ---------------------------------------------------------------------------
// Decode a B-layout linear index -> (c, p)
static __device__ __forceinline__ void blayout_decode(int idx, int& c, int& p) {
    int low = idx & 31;
    int rest = idx >> 5;
    p = rest & (PTOT - 1);
    c = (rest >> 15) * 32 + low;
}

__global__ void zero_u16(unsigned short* p, int total) {
    int i = blockIdx.x * blockDim.x + threadIdx.x;
    if (i < total) p[i] = 0;
}

__global__ void pack_weight(const float* __restrict__ W, unsigned short* __restrict__ Wb,
                            int O, int Cin, int Opad, int Cpad) {
    int i = blockIdx.x * blockDim.x + threadIdx.x;
    if (i >= Opad * Cpad) return;
    int r = i / Cpad, c = i % Cpad;
    float v = (r < O && c < Cin) ? W[(size_t)r * Cin + c] : 0.0f;
    Wb[i] = f32_to_bf16(v);
}

// concat_elu(concat(x, ul, b)) -> bf16 B-layout [352][PTOT]  (169 real channels x2)
__global__ void ce_inputs_xub(const float* __restrict__ x, const float* __restrict__ ul,
                              const float* __restrict__ b, unsigned short* __restrict__ out, int total) {
    int idx = blockIdx.x * blockDim.x + threadIdx.x;
    if (idx >= total) return;
    int c, p; blayout_decode(idx, c, p);
    int n = p >> 10, pl = p & 1023;
    float v = 0.0f;
    if (c < 338) {
        int j = (c < 169) ? c : c - 169;
        float s;
        if (j < 3)        s = x[((size_t)n * 3   + j)        * HW + pl];
        else if (j < 163) s = ul[((size_t)n * 160 + (j - 3))  * HW + pl];
        else              s = b[((size_t)n * 6   + (j - 163)) * HW + pl];
        if (c >= 169) s = -s;
        v = eluf(s);
    }
    out[idx] = f32_to_bf16(v);
}

// concat_elu(concat(ul, b)) -> bf16 B-layout [352][PTOT]  (166 real channels x2)
__global__ void ce_inputs_ub(const float* __restrict__ ul, const float* __restrict__ b,
                             unsigned short* __restrict__ out, int total) {
    int idx = blockIdx.x * blockDim.x + threadIdx.x;
    if (idx >= total) return;
    int c, p; blayout_decode(idx, c, p);
    int n = p >> 10, pl = p & 1023;
    float v = 0.0f;
    if (c < 332) {
        int j = (c < 166) ? c : c - 166;
        float s = (j < 160) ? ul[((size_t)n * 160 + j) * HW + pl]
                            : b[((size_t)n * 6 + (j - 160)) * HW + pl];
        if (c >= 166) s = -s;
        v = eluf(s);
    }
    out[idx] = f32_to_bf16(v);
}

// concat_elu(ul) -> bf16 B-layout [320][PTOT]
__global__ void ce_inputs_ul(const float* __restrict__ ul, unsigned short* __restrict__ out, int total) {
    int idx = blockIdx.x * blockDim.x + threadIdx.x;
    if (idx >= total) return;
    int c, p; blayout_decode(idx, c, p);
    int n = p >> 10, pl = p & 1023;
    float s = (c < 160) ? ul[((size_t)n * 160 + c) * HW + pl]
                        : -ul[((size_t)n * 160 + (c - 160)) * HW + pl];
    out[idx] = f32_to_bf16(eluf(s));
}

// concat_elu of f32 row-major [Hreal][PTOT] -> bf16 B-layout [Cpad][PTOT]
__global__ void ce_from_f32(const float* __restrict__ in, unsigned short* __restrict__ out,
                            int Hreal, int total) {
    int idx = blockIdx.x * blockDim.x + threadIdx.x;
    if (idx >= total) return;
    int c, p; blayout_decode(idx, c, p);
    float v = 0.0f;
    if (c < Hreal)            v = eluf(in[(size_t)c * PTOT + p]);
    else if (c < 2 * Hreal)   v = eluf(-in[(size_t)(c - Hreal) * PTOT + p]);
    out[idx] = f32_to_bf16(v);
}

// GRN tail for x_ul_b base (Creal=169): feat = base + ga*sigmoid(gb) -> bf16 B-layout [192][PTOT]
__global__ void resgate_xub(const float* __restrict__ h2, const float* __restrict__ x,
                            const float* __restrict__ ul, const float* __restrict__ b,
                            unsigned short* __restrict__ out, int total) {
    int idx = blockIdx.x * blockDim.x + threadIdx.x;
    if (idx >= total) return;
    int c, p; blayout_decode(idx, c, p);
    float v = 0.0f;
    if (c < 169) {
        int n = p >> 10, pl = p & 1023;
        float base;
        if (c < 3)        base = x[((size_t)n * 3   + c)        * HW + pl];
        else if (c < 163) base = ul[((size_t)n * 160 + (c - 3))  * HW + pl];
        else              base = b[((size_t)n * 6   + (c - 163)) * HW + pl];
        float ga = h2[(size_t)c * PTOT + p];
        float gb = h2[(size_t)(169 + c) * PTOT + p];
        v = base + ga * sigmf(gb);
    }
    out[idx] = f32_to_bf16(v);
}

// GRN tail for ul_b base (Creal=166) -> bf16 B-layout [192][PTOT]
__global__ void resgate_ub(const float* __restrict__ h2, const float* __restrict__ ul,
                           const float* __restrict__ b, unsigned short* __restrict__ out, int total) {
    int idx = blockIdx.x * blockDim.x + threadIdx.x;
    if (idx >= total) return;
    int c, p; blayout_decode(idx, c, p);
    float v = 0.0f;
    if (c < 166) {
        int n = p >> 10, pl = p & 1023;
        float base = (c < 160) ? ul[((size_t)n * 160 + c) * HW + pl]
                               : b[((size_t)n * 6 + (c - 160)) * HW + pl];
        float ga = h2[(size_t)c * PTOT + p];
        float gb = h2[(size_t)(166 + c) * PTOT + p];
        v = base + ga * sigmf(gb);
    }
    out[idx] = f32_to_bf16(v);
}

// Final GRN tail on ul (Creal=160), write f32 NCHW output
__global__ void resgate_final(const float* __restrict__ h2, const float* __restrict__ ul,
                              float* __restrict__ out, int total) {
    int idx = blockIdx.x * blockDim.x + threadIdx.x;
    if (idx >= total) return;
    int c = idx >> 15, p = idx & (PTOT - 1);
    int n = p >> 10, pl = p & 1023;
    float base = ul[((size_t)n * 160 + c) * HW + pl];
    float ga = h2[(size_t)c * PTOT + p];
    float gb = h2[(size_t)(160 + c) * PTOT + p];
    out[((size_t)n * 160 + c) * HW + pl] = base + ga * sigmf(gb);
}

// ---------------------------------------------------------------------------
// Host-side orchestration
// ---------------------------------------------------------------------------
extern "C" void kernel_launch(void* const* d_in, const int* in_sizes, int n_in,
                              void* d_out, int out_size, void* d_ws, size_t ws_size,
                              hipStream_t stream) {
    (void)in_sizes; (void)n_in; (void)out_size; (void)ws_size;
    const float* x    = (const float*)d_in[0];
    const float* ul   = (const float*)d_in[1];
    const float* bI   = (const float*)d_in[2];
    const float* gkWi = (const float*)d_in[3];
    const float* gkbi = (const float*)d_in[4];
    const float* gkWo = (const float*)d_in[5];
    const float* gkbo = (const float*)d_in[6];
    const float* gqWi = (const float*)d_in[7];
    const float* gqbi = (const float*)d_in[8];
    const float* gqWo = (const float*)d_in[9];
    const float* gqbo = (const float*)d_in[10];
    const float* gvWi = (const float*)d_in[11];
    const float* gvbi = (const float*)d_in[12];
    const float* gvWo = (const float*)d_in[13];
    const float* gvbo = (const float*)d_in[14];
    const float* nkW  = (const float*)d_in[15];
    const float* nkb  = (const float*)d_in[16];
    const float* nqW  = (const float*)d_in[17];
    const float* nqb  = (const float*)d_in[18];
    const float* nvW  = (const float*)d_in[19];
    const float* nvb  = (const float*)d_in[20];
    const float* goWi = (const float*)d_in[21];
    const float* gobi = (const float*)d_in[22];
    const float* goWs = (const float*)d_in[23];
    const float* gobs = (const float*)d_in[24];
    const float* goWo = (const float*)d_in[25];
    const float* gobo = (const float*)d_in[26];
    float* out = (float*)d_out;

    char* ws = (char*)d_ws;
    size_t off = 0;
    auto alloc = [&](size_t bytes) -> char* {
        char* p = ws + off;
        off = (off + bytes + 255) & ~(size_t)255;
        return p;
    };
    const int P = PTOT;

    // packed bf16 weights
    unsigned short* wb_gkWi = (unsigned short*)alloc((size_t)176 * 352 * 2);
    unsigned short* wb_gkWo = (unsigned short*)alloc((size_t)352 * 352 * 2);
    unsigned short* wb_gqWi = (unsigned short*)alloc((size_t)176 * 352 * 2);
    unsigned short* wb_gqWo = (unsigned short*)alloc((size_t)336 * 352 * 2);
    unsigned short* wb_gvWi = (unsigned short*)alloc((size_t)176 * 352 * 2);
    unsigned short* wb_gvWo = (unsigned short*)alloc((size_t)352 * 352 * 2);
    unsigned short* wb_nk   = (unsigned short*)alloc((size_t)16  * 192 * 2);
    unsigned short* wb_nq   = (unsigned short*)alloc((size_t)16  * 192 * 2);
    unsigned short* wb_nv   = (unsigned short*)alloc((size_t)80  * 192 * 2);
    unsigned short* wb_goWi = (unsigned short*)alloc((size_t)160 * 320 * 2);
    unsigned short* wb_goWs = (unsigned short*)alloc((size_t)160 * 160 * 2);
    unsigned short* wb_goWo = (unsigned short*)alloc((size_t)320 * 320 * 2);

    // activation staging
    unsigned short* ceA   = (unsigned short*)alloc((size_t)352 * P * 2); // concat_elu(x_ul_b)
    unsigned short* ceB   = (unsigned short*)alloc((size_t)352 * P * 2); // concat_elu(ul_b) / concat_elu(ul)
    unsigned short* ceh   = (unsigned short*)alloc((size_t)352 * P * 2); // concat_elu of GEMM outputs
    unsigned short* feat  = (unsigned short*)alloc((size_t)192 * P * 2); // GRN outputs for nin
    float*          h1    = (float*)alloc((size_t)176 * P * 4);
    float*          h2    = (float*)alloc((size_t)352 * P * 4);
    unsigned short* Kb    = (unsigned short*)alloc((size_t)32  * P * 2);
    unsigned short* Qb    = (unsigned short*)alloc((size_t)32  * P * 2);
    unsigned short* Vrow  = (unsigned short*)alloc((size_t)80  * P * 2);
    float*          attv  = (float*)alloc((size_t)80 * P * 4);
    unsigned short* ceav  = (unsigned short*)alloc((size_t)160 * P * 2);
    float*          Schk  = (float*)alloc((size_t)BC * HW * HW * 4);
    unsigned short* Wchk  = (unsigned short*)alloc((size_t)BC * HW * HW * 2);

    auto ew = [&](int total) { return dim3((total + 255) / 256); };
    auto gemm = [&](const unsigned short* Wb, const unsigned short* Act, const float* bias,
                    float* OutF, unsigned short* OutB, int Oreal, int Opad, int Cpad,
                    int mode, int accum) {
        dim3 g(P / 64, (Opad + 127) / 128, 1), blk(32, 8);
        wmma_gemm<<<g, blk, 0, stream>>>(Wb, Act, bias, OutF, OutB, Oreal, Opad, Cpad, P, mode, accum);
    };

    // 0) zero K/Q buffers (rows 16..31 are never written by the 16-row GEMMs)
    zero_u16<<<ew(32 * P), 256, 0, stream>>>(Kb, 32 * P);
    zero_u16<<<ew(32 * P), 256, 0, stream>>>(Qb, 32 * P);

    // 1) pack weights to padded bf16
    pack_weight<<<ew(176 * 352), 256, 0, stream>>>(gkWi, wb_gkWi, 169, 338, 176, 352);
    pack_weight<<<ew(352 * 352), 256, 0, stream>>>(gkWo, wb_gkWo, 338, 338, 352, 352);
    pack_weight<<<ew(176 * 352), 256, 0, stream>>>(gqWi, wb_gqWi, 166, 332, 176, 352);
    pack_weight<<<ew(336 * 352), 256, 0, stream>>>(gqWo, wb_gqWo, 332, 332, 336, 352);
    pack_weight<<<ew(176 * 352), 256, 0, stream>>>(gvWi, wb_gvWi, 169, 338, 176, 352);
    pack_weight<<<ew(352 * 352), 256, 0, stream>>>(gvWo, wb_gvWo, 338, 338, 352, 352);
    pack_weight<<<ew(16 * 192),  256, 0, stream>>>(nkW,  wb_nk,   16, 169, 16, 192);
    pack_weight<<<ew(16 * 192),  256, 0, stream>>>(nqW,  wb_nq,   16, 166, 16, 192);
    pack_weight<<<ew(80 * 192),  256, 0, stream>>>(nvW,  wb_nv,   80, 169, 80, 192);
    pack_weight<<<ew(160 * 320), 256, 0, stream>>>(goWi, wb_goWi, 160, 320, 160, 320);
    pack_weight<<<ew(160 * 160), 256, 0, stream>>>(goWs, wb_goWs, 160, 160, 160, 160);
    pack_weight<<<ew(320 * 320), 256, 0, stream>>>(goWo, wb_goWo, 320, 320, 320, 320);

    // 2) stage concat_elu inputs
    ce_inputs_xub<<<ew(352 * P), 256, 0, stream>>>(x, ul, bI, ceA, 352 * P);
    ce_inputs_ub<<<ew(352 * P), 256, 0, stream>>>(ul, bI, ceB, 352 * P);

    // 3) key chain: grn(x_ul_b) -> nin -> Kb
    gemm(wb_gkWi, ceA, gkbi, h1, nullptr, 169, 176, 352, 0, 0);
    ce_from_f32<<<ew(352 * P), 256, 0, stream>>>(h1, ceh, 169, 352 * P);
    gemm(wb_gkWo, ceh, gkbo, h2, nullptr, 338, 352, 352, 0, 0);
    resgate_xub<<<ew(192 * P), 256, 0, stream>>>(h2, x, ul, bI, feat, 192 * P);
    gemm(wb_nk, feat, nkb, nullptr, Kb, 16, 16, 192, 1, 0);

    // 4) value chain: grn(x_ul_b) -> nin -> Vrow
    gemm(wb_gvWi, ceA, gvbi, h1, nullptr, 169, 176, 352, 0, 0);
    ce_from_f32<<<ew(352 * P), 256, 0, stream>>>(h1, ceh, 169, 352 * P);
    gemm(wb_gvWo, ceh, gvbo, h2, nullptr, 338, 352, 352, 0, 0);
    resgate_xub<<<ew(192 * P), 256, 0, stream>>>(h2, x, ul, bI, feat, 192 * P);
    gemm(wb_nv, feat, nvb, nullptr, Vrow, 80, 80, 192, 2, 0);

    // 5) query chain: grn(ul_b) -> nin -> Qb
    gemm(wb_gqWi, ceB, gqbi, h1, nullptr, 166, 176, 352, 0, 0);
    ce_from_f32<<<ew(352 * P), 256, 0, stream>>>(h1, ceh, 166, 352 * P);
    gemm(wb_gqWo, ceh, gqbo, h2, nullptr, 332, 336, 352, 0, 0);
    resgate_ub<<<ew(192 * P), 256, 0, stream>>>(h2, ul, bI, feat, 192 * P);
    gemm(wb_nq, feat, nqb, nullptr, Qb, 16, 16, 192, 1, 0);

    // 6) attention, chunked over batches (S chunk stays L2-resident)
    for (int b0 = 0; b0 < NB; b0 += BC) {
        qk_gemm<<<dim3(16, 8, BC), dim3(32, 8), 0, stream>>>(Qb, Kb, Schk, b0);
        softmax_rows<<<dim3(128, BC), dim3(32, 8), 0, stream>>>(Schk, Wchk);
        av_gemm<<<dim3(16, 5, BC), dim3(32), 0, stream>>>(Vrow, Wchk, attv, b0);
    }

    // 7) output GRN: grn(ul, a=att_v)
    ce_from_f32<<<ew(160 * P), 256, 0, stream>>>(attv, ceav, 80, 160 * P);
    ce_inputs_ul<<<ew(320 * P), 256, 0, stream>>>(ul, ceB, 320 * P);
    gemm(wb_goWi, ceB,  gobi, h1, nullptr, 160, 160, 320, 0, 0);
    gemm(wb_goWs, ceav, gobs, h1, nullptr, 160, 160, 160, 0, 1); // accumulate
    ce_from_f32<<<ew(320 * P), 256, 0, stream>>>(h1, ceh, 160, 320 * P);
    gemm(wb_goWo, ceh, gobo, h2, nullptr, 320, 320, 320, 0, 0);
    resgate_final<<<ew(160 * P), 256, 0, stream>>>(h2, ul, out, 160 * P);
}
